// FlashAttention3Core_51110110822810
// MI455X (gfx1250) — matrix operand order repaired and checked by reading, not compiled, not run on hardware
//
#include <hip/hip_runtime.h>

// ---------------------------------------------------------------------------
// FlashAttention forward for MI455X (gfx1250, wave32, WMMA bf16 + TDM).
// B=2, H=16, N=S=2048, D=128. fp32 in/out; bf16 WMMA with f32 accumulation.
// K/V tiles staged into LDS by the Tensor Data Mover, double-buffered.
// ---------------------------------------------------------------------------

typedef __bf16 bf16;
typedef __attribute__((ext_vector_type(16))) bf16  v16bf;
typedef __attribute__((ext_vector_type(8)))  bf16  v8bf;
typedef __attribute__((ext_vector_type(8)))  float v8f;
typedef __attribute__((ext_vector_type(4)))  unsigned int v4u;
typedef __attribute__((ext_vector_type(4)))  int   v4i;
typedef __attribute__((ext_vector_type(8)))  int   v8i;

constexpr int Bc = 2, Hc = 16, Nc = 2048, Sc = 2048, Dc = 128;
constexpr size_t NELEM = (size_t)Bc * Hc * Nc * Dc;   // 8,388,608 (same for K,V)

// LDS map (single static block -> descriptor lds_addr offsets are compile-time)
constexpr unsigned KV_BUF_BYTES = 64 * 128 * 2;            // 16 KB per tile
constexpr unsigned LDS_K_OFF = 0;                          // 2 buffers: 32 KB
constexpr unsigned LDS_V_OFF = 2 * KV_BUF_BYTES;           // 2 buffers: 32 KB
constexpr unsigned LDS_P_OFF = 4 * KV_BUF_BYTES;           // 8*16*72*2 = 18 KB
constexpr unsigned LDS_TOTAL = LDS_P_OFF + 8 * 16 * 72 * 2; // 83,968 B

// ---- lane reductions within 16-lane halves (rows live per half-wave) ------
template <int MASK>
__device__ __forceinline__ float swz_xor(float x) {
  int i = __builtin_amdgcn_ds_swizzle(__float_as_int(x), (MASK << 10) | 0x1f);
  return __int_as_float(i);
}
__device__ __forceinline__ float redmax16(float x) {
  x = fmaxf(x, swz_xor<1>(x));
  x = fmaxf(x, swz_xor<2>(x));
  x = fmaxf(x, swz_xor<4>(x));
  x = fmaxf(x, swz_xor<8>(x));
  return x;
}
__device__ __forceinline__ float redsum16(float x) {
  x += swz_xor<1>(x);
  x += swz_xor<2>(x);
  x += swz_xor<4>(x);
  x += swz_xor<8>(x);
  return x;
}

// ---- Tensor Data Mover: 2-D tile load, global -> LDS ----------------------
// Builds D# group0/group1 per CDNA5 ISA and issues the
// 6-arg (clang-23 / therock) builtin. Dims/strides in data_size (=2B) units.
__device__ __forceinline__ void tdm_load_2d(unsigned lds_byte_off,
                                            const void* gptr,
                                            unsigned tensor_d0, unsigned tensor_d1,
                                            unsigned tile_d0, unsigned tile_d1,
                                            unsigned d0_stride) {
  unsigned long long ga = (unsigned long long)gptr;  // byte address
  v4u g0;
  g0[0] = 1u;                                    // count=1 (valid user D#)
  g0[1] = lds_byte_off;                          // lds_addr [63:32]
  g0[2] = (unsigned)(ga & 0xffffffffu);          // global_addr [95:64]
  g0[3] = (unsigned)((ga >> 32) & 0x01ffffffu)   // global_addr [120:96]
          | (2u << 30);                          // type=2 ("image") [127:126]
  v8i g1;
  g1[0] = (int)(1u << 16);                       // data_size=1 (2 bytes)
  g1[1] = (int)(tensor_d0 << 16);                // tensor_dim0[15:0] @ [63:48]
  g1[2] = (int)((tensor_d0 >> 16) | (tensor_d1 << 16)); // dim0 hi | dim1 lo
  g1[3] = (int)((tensor_d1 >> 16) | (tile_d0 << 16));   // dim1 hi | tile_dim0
  g1[4] = (int)(tile_d1 & 0xffffu);              // tile_dim1 (tile_dim2=0: 2D)
  g1[5] = (int)d0_stride;                        // tensor_dim0_stride[31:0]
  g1[6] = 0;                                     // stride0 hi | dim1_stride lo
  g1[7] = 0;
  v4i z4 = {0, 0, 0, 0};                         // groups 2/3 unused (2-D)
  v8i z8 = {0, 0, 0, 0, 0, 0, 0, 0};
  __builtin_amdgcn_tensor_load_to_lds(g0, g1, z4, z4, z8, 0);
}

// ---- prepass: fp32 -> bf16 (Q, K) -----------------------------------------
__global__ __launch_bounds__(256) void cvt_bf16_kernel(const float* __restrict__ in,
                                                       bf16* __restrict__ out) {
  size_t i = (size_t)blockIdx.x * blockDim.x + threadIdx.x;
  if (i < NELEM) out[i] = (bf16)in[i];
}

// ---- prepass: V [bh, S, D] fp32 -> Vt [bh, D, S] bf16 ---------------------
__global__ __launch_bounds__(256) void transpose_v_kernel(const float* __restrict__ v,
                                                          bf16* __restrict__ vt) {
  size_t i = (size_t)blockIdx.x * blockDim.x + threadIdx.x;  // output-linear
  if (i >= NELEM) return;
  constexpr size_t DS = (size_t)Dc * Sc;
  size_t bh = i / DS;
  size_t rem = i - bh * DS;
  size_t d = rem / Sc;
  size_t s = rem - d * Sc;
  vt[i] = (bf16)v[(bh * Sc + s) * Dc + d];
}

// ---- main flash attention kernel ------------------------------------------
// Grid: (N/128, B*H). Block: 256 threads = 8 waves; wave w owns 16 Q rows.
__global__ __launch_bounds__(256) void fa_fwd_kernel(const bf16* __restrict__ Qb,
                                                     const bf16* __restrict__ Kb,
                                                     const bf16* __restrict__ Vt,
                                                     float* __restrict__ Out) {
  __shared__ __align__(128) unsigned char smem[LDS_TOTAL];

  const int wave = threadIdx.x >> 5;
  const int lane = threadIdx.x & 31;
  const int col  = lane & 15;       // N-index inside a 16-wide tile
  const int hi   = lane >> 4;       // 0: lanes 0-15, 1: lanes 16-31
  const int bh   = blockIdx.y;
  const int qbase = blockIdx.x * 128 + wave * 16;

  const bf16* kg = Kb + (size_t)bh * Sc * Dc;               // K [S, D]
  const bf16* vg = Vt + (size_t)bh * (size_t)Dc * Sc;       // Vt [D, S]

  // --- load Q A-fragments: 16x128 = 4 frags of 16x32 (A layout per ISA) ----
  v16bf aq[4];
  {
    const bf16* qrow = Qb + ((size_t)bh * Nc + qbase + col) * Dc;
    const int ko = hi ? 8 : 0;      // lanes 16-31 hold K = {8..15, 24..31}
#pragma unroll
    for (int j = 0; j < 4; ++j) {
      v8bf lo = *(const v8bf*)(qrow + 32 * j + ko);
      v8bf hh = *(const v8bf*)(qrow + 32 * j + 16 + ko);
      aq[j] = __builtin_shufflevector(lo, hh, 0, 1, 2, 3, 4, 5, 6, 7,
                                      8, 9, 10, 11, 12, 13, 14, 15);
    }
  }

  // --- accumulators & row stats --------------------------------------------
  v8f o_acc[8];                     // O tile 16x128: 8 N-tiles of 16x16 f32
#pragma unroll
  for (int dt = 0; dt < 8; ++dt)
#pragma unroll
    for (int i = 0; i < 8; ++i) o_acc[dt][i] = 0.0f;

  float m_r[8], l_r[8];
#pragma unroll
  for (int r = 0; r < 8; ++r) { m_r[r] = -3.0e38f; l_r[r] = 0.0f; }

  const float SC = 0.08838834764831845f * 1.4426950408889634f; // rsqrt(128)*log2(e)

  // --- TDM prologue: stage tile 0 into buffer 0 ----------------------------
  if (wave == 0) {
    // K tile: 64 rows (keys) x 128 cols (D); contiguous dim = D
    tdm_load_2d(LDS_K_OFF, kg, Dc, Sc, Dc, 64, Dc);
    // V tile: 128 rows (D) x 64 cols (keys); contiguous dim = S
    tdm_load_2d(LDS_V_OFF, vg, Sc, Dc, 64, Dc, Sc);
  }

  for (int it = 0; it < Sc / 64; ++it) {
    const int s0  = it * 64;
    const int buf = it & 1;

    // All waves are done reading buffer buf^1 (read in the previous iteration).
    __syncthreads();
    if (wave == 0) {
      const int s0n = (s0 + 64 < Sc) ? (s0 + 64) : s0;   // clamp tail prefetch
      tdm_load_2d(LDS_K_OFF + (buf ^ 1) * KV_BUF_BYTES,
                  kg + (size_t)s0n * Dc, Dc, Sc, Dc, 64, Dc);
      tdm_load_2d(LDS_V_OFF + (buf ^ 1) * KV_BUF_BYTES,
                  vg + s0n, Sc, Dc, 64, Dc, Sc);
      // TDM ops from one wave complete in order: <=2 outstanding leaves only
      // the two just-issued prefetches in flight, i.e. tile `it` has landed.
      __builtin_amdgcn_s_wait_tensorcnt(2);
    }
    __syncthreads();   // publish tile `it` to all waves

    const bf16* kt  = (const bf16*)(smem + LDS_K_OFF + buf * KV_BUF_BYTES); // [64][128]
    const bf16* vtt = (const bf16*)(smem + LDS_V_OFF + buf * KV_BUF_BYTES); // [128][64]

    // ---- S = Q * K^T : 4 tiles of 16x16, K-dim = D = 128 (4 wmma each) ----
    v8f sacc[4];
#pragma unroll
    for (int t = 0; t < 4; ++t)
#pragma unroll
      for (int i = 0; i < 8; ++i) sacc[t][i] = 0.0f;

#pragma unroll
    for (int t = 0; t < 4; ++t) {
      // B layout: N = col (key index), K = {8h..8h+7, 16+8h..16+8h+7}
      const bf16* krow = kt + (t * 16 + col) * Dc + (hi ? 8 : 0);
#pragma unroll
      for (int ks = 0; ks < 4; ++ks) {
        v8bf blo = *(const v8bf*)(krow + ks * 32);
        v8bf bhi = *(const v8bf*)(krow + ks * 32 + 16);
        v16bf bk = __builtin_shufflevector(blo, bhi, 0, 1, 2, 3, 4, 5, 6, 7,
                                           8, 9, 10, 11, 12, 13, 14, 15);
        sacc[t] = __builtin_amdgcn_wmma_f32_16x16x32_bf16(
            false, aq[ks], false, bk, (short)0, sacc[t], false, false);
      }
    }

    // ---- online softmax over the 16x64 logit tile -------------------------
    bf16* pw = (bf16*)(smem + LDS_P_OFF) + wave * 16 * 72;  // wave-private
#pragma unroll
    for (int r = 0; r < 8; ++r) {
      float s0v = sacc[0][r] * SC;
      float s1v = sacc[1][r] * SC;
      float s2v = sacc[2][r] * SC;
      float s3v = sacc[3][r] * SC;
      float mx = redmax16(fmaxf(fmaxf(s0v, s1v), fmaxf(s2v, s3v)));
      float mnew = fmaxf(m_r[r], mx);
      float corr = __builtin_amdgcn_exp2f(m_r[r] - mnew);
      float p0 = __builtin_amdgcn_exp2f(s0v - mnew);
      float p1 = __builtin_amdgcn_exp2f(s1v - mnew);
      float p2 = __builtin_amdgcn_exp2f(s2v - mnew);
      float p3 = __builtin_amdgcn_exp2f(s3v - mnew);

      const int row = r + (hi ? 8 : 0);     // C-layout row for this VGPR
      pw[row * 72 +  0 + col] = (bf16)p0;
      pw[row * 72 + 16 + col] = (bf16)p1;
      pw[row * 72 + 32 + col] = (bf16)p2;
      pw[row * 72 + 48 + col] = (bf16)p3;

      l_r[r] = l_r[r] * corr + redsum16(p0 + p1 + p2 + p3);
      m_r[r] = mnew;
#pragma unroll
      for (int dt = 0; dt < 8; ++dt) o_acc[dt][r] *= corr;
    }

    // ---- re-fetch P as bf16 A-fragments (16x64 -> 2 frags of 16x32) ------
    // Same-wave LDS ops are in-order; region is wave-private (no barrier).
    v16bf pa[2];
    {
      const bf16* prow = pw + col * 72;
      const int ko = hi ? 8 : 0;
#pragma unroll
      for (int kk = 0; kk < 2; ++kk) {
        v8bf lo = *(const v8bf*)(prow + kk * 32 + ko);
        v8bf hh = *(const v8bf*)(prow + kk * 32 + 16 + ko);
        pa[kk] = __builtin_shufflevector(lo, hh, 0, 1, 2, 3, 4, 5, 6, 7,
                                         8, 9, 10, 11, 12, 13, 14, 15);
      }
    }

    // ---- O += P * V : 8 D-tiles, K-dim = 64 keys (2 wmma each) -----------
#pragma unroll
    for (int dt = 0; dt < 8; ++dt) {
      // Vt tile is [D][64 keys]: N = output column d (= col), K = {8h.., 16+8h..}
      const bf16* vrow = vtt + (dt * 16 + col) * 64 + (hi ? 8 : 0);
#pragma unroll
      for (int kk = 0; kk < 2; ++kk) {
        v8bf vlo = *(const v8bf*)(vrow + kk * 32);
        v8bf vhi = *(const v8bf*)(vrow + kk * 32 + 16);
        v16bf bv = __builtin_shufflevector(vlo, vhi, 0, 1, 2, 3, 4, 5, 6, 7,
                                           8, 9, 10, 11, 12, 13, 14, 15);
        o_acc[dt] = __builtin_amdgcn_wmma_f32_16x16x32_bf16(
            false, pa[kk], false, bv, (short)0, o_acc[dt], false, false);
      }
    }
  }

  // ---- epilogue: O / l ----------------------------------------------------
#pragma unroll
  for (int r = 0; r < 8; ++r) {
    const float inv = 1.0f / l_r[r];
    const int row = qbase + r + (hi ? 8 : 0);
    float* orow = Out + ((size_t)bh * Nc + row) * Dc + col;
#pragma unroll
    for (int dt = 0; dt < 8; ++dt) orow[dt * 16] = o_acc[dt][r] * inv;
  }
}

// ---------------------------------------------------------------------------
extern "C" void kernel_launch(void* const* d_in, const int* in_sizes, int n_in,
                              void* d_out, int out_size, void* d_ws, size_t ws_size,
                              hipStream_t stream) {
  (void)in_sizes; (void)n_in; (void)out_size; (void)ws_size;
  const float* q = (const float*)d_in[0];
  const float* k = (const float*)d_in[1];
  const float* v = (const float*)d_in[2];
  float* out = (float*)d_out;

  bf16* Qb = (bf16*)d_ws;          // 16.78 MB
  bf16* Kb = Qb + NELEM;           // 16.78 MB
  bf16* Vt = Kb + NELEM;           // 16.78 MB (transposed [bh, D, S])

  const int cvt_blocks = (int)((NELEM + 255) / 256);
  cvt_bf16_kernel<<<cvt_blocks, 256, 0, stream>>>(q, Qb);
  cvt_bf16_kernel<<<cvt_blocks, 256, 0, stream>>>(k, Kb);
  transpose_v_kernel<<<cvt_blocks, 256, 0, stream>>>(v, Vt);

  dim3 grid(Nc / 128, Bc * Hc);    // (16, 32)
  fa_fwd_kernel<<<grid, 256, 0, stream>>>(Qb, Kb, Vt, out);
}
